// RBFNetCV_62998580298060
// MI455X (gfx1250) — matrix operand order repaired and checked by reading, not compiled, not run on hardware
//
#include <hip/hip_runtime.h>

#define NROWS 8192
#define KCENT 4096
#define DDIM  256
#define LPITCH 264   // LDS row pitch (halves): 256 + 8 pad -> conflict-free ds_load_b128

typedef __attribute__((ext_vector_type(16))) _Float16 v16h;
typedef __attribute__((ext_vector_type(8)))  _Float16 v8h;
typedef __attribute__((ext_vector_type(8)))  float    v8f;

#if __has_builtin(__builtin_amdgcn_global_load_async_to_lds_b128)
#define ASYNC_LDS 1
typedef int v4i __attribute__((vector_size(16)));
typedef __attribute__((address_space(1))) v4i* gp128;   // global source (AS1)
typedef __attribute__((address_space(3))) v4i* lp128;   // LDS destination (AS3)
#else
#define ASYNC_LDS 0
#endif

static __device__ __forceinline__ v8f wmma16(v16h a, v16h b, v8f c) {
  // D = A(16x32 f16) * B(32x16 f16) + C(16x16 f32)
  return __builtin_amdgcn_wmma_f32_16x16x32_f16(
      false, a, false, b, (short)0, c, false, false);
}

static __device__ __forceinline__ v16h join16(v8h a, v8h b) {
  return __builtin_shufflevector(a, b, 0, 1, 2, 3, 4, 5, 6, 7,
                                 8, 9, 10, 11, 12, 13, 14, 15);
}

static __device__ __forceinline__ float fast_exp2(float x) {
#if __has_builtin(__builtin_amdgcn_exp2f)
  return __builtin_amdgcn_exp2f(x);   // v_exp_f32
#else
  return exp2f(x);
#endif
}

// Split f32 into f16 hi + f16 lo (residual): ~22-bit effective mantissa.
__global__ void k_split(const float* __restrict__ src, _Float16* __restrict__ hi,
                        _Float16* __restrict__ lo, int n) {
  int i = blockIdx.x * blockDim.x + threadIdx.x;
  if (i < n) {
    float v = src[i];
    _Float16 h = (_Float16)v;
    hi[i] = h;
    lo[i] = (_Float16)(v - (float)h);
  }
}

// ex[r] = exp(-sigma * ||x_r||^2)
__global__ void k_prep_x(const float* __restrict__ src, float* __restrict__ ex,
                         const float* __restrict__ sigp, int rows) {
  int r = blockIdx.x * blockDim.x + threadIdx.x;
  if (r < rows) {
    float sg = *sigp;
    const float4* p = (const float4*)(src + (size_t)r * DDIM);
    float s = 0.f;
#pragma unroll 8
    for (int i = 0; i < DDIM / 4; ++i) {
      float4 v = p[i];
      s += v.x * v.x + v.y * v.y + v.z * v.z + v.w * v.w;
    }
    ex[r] = __expf(-sg * s);
  }
}

// we[r] = w[r] * exp(-sigma * ||c_r||^2)
__global__ void k_prep_c(const float* __restrict__ src, const float* __restrict__ w,
                         float* __restrict__ we, const float* __restrict__ sigp,
                         int rows) {
  int r = blockIdx.x * blockDim.x + threadIdx.x;
  if (r < rows) {
    float sg = *sigp;
    const float4* p = (const float4*)(src + (size_t)r * DDIM);
    float s = 0.f;
#pragma unroll 8
    for (int i = 0; i < DDIM / 4; ++i) {
      float4 v = p[i];
      s += v.x * v.x + v.y * v.y + v.z * v.z + v.w * v.w;
    }
    we[r] = w[r] * __expf(-sg * s);
  }
}

// out[i] = ex[i] * sum_p part[p][i]   (fixed order -> deterministic)
__global__ void k_reduce(const float* __restrict__ part, const float* __restrict__ ex,
                         float* __restrict__ out, int n, int nparts) {
  int i = blockIdx.x * blockDim.x + threadIdx.x;
  if (i < n) {
    float s = 0.f;
    for (int p = 0; p < nparts; ++p) s += part[(size_t)p * n + i];
    out[i] = ex[i] * s;
  }
}

// Main fused kernel: per block 64 rows x 512 centers (8 tiles of 64 centers).
// B (centers) staged in LDS via async global->LDS; shared by all 4 waves.
__global__ __launch_bounds__(128)
void k_rbf(const _Float16* __restrict__ xh, const _Float16* __restrict__ xl,
           const _Float16* __restrict__ ch, const _Float16* __restrict__ cl,
           const float* __restrict__ we, const float* __restrict__ sigp,
           float* __restrict__ part) {
  __shared__ _Float16 sBh[64 * LPITCH];
  __shared__ _Float16 sBl[64 * LPITCH];

  const int tid  = threadIdx.x;
  const int lane = tid & 31;
  const int wave = tid >> 5;
  const int nl   = lane & 15;          // column / row-in-half index
  const int hiH  = (lane >> 4) & 1;    // which 16-lane half
  const int m0   = blockIdx.x * 64 + wave * 16;
  const int cb0  = blockIdx.y * 512;
  const float s2 = 2.0f * (*sigp) * 1.4426950408889634f;  // 2*sigma*log2(e)

  // A fragment addressing (16-bit A 16x32 layout):
  //   lanes 0-15 : row M=lane,    K = kk+{0..7}  and kk+{16..23}
  //   lanes 16-31: row M=lane-16, K = kk+{8..15} and kk+{24..31}
  const _Float16* xh_row = xh + (size_t)(m0 + nl) * DDIM;
  const _Float16* xl_row = xl + (size_t)(m0 + nl) * DDIM;
  const int aoff = hiH * 8;
  const int bofs = hiH * 8;            // B: lanes 16-31 hold K = kk+{8..15} and kk+{24..31}

  float acc_out[8];
#pragma unroll
  for (int j = 0; j < 8; ++j) acc_out[j] = 0.f;

  for (int ct = 0; ct < 8; ++ct) {
    const int cbase = cb0 + ct * 64;
    if (ct) __syncthreads();           // previous tile's ds reads done

    // Stage 64x256 hi/lo center tile: contiguous 32 KB per plane in global.
    {
      const _Float16* gh = ch + (size_t)cbase * DDIM;
      const _Float16* gl = cl + (size_t)cbase * DDIM;
#pragma unroll
      for (int i = 0; i < 16; ++i) {
        int chunk = i * 128 + tid;     // 2048 chunks of 8 halves (16 B)
        int r  = chunk >> 5;
        int cc = chunk & 31;
        int lofs = r * LPITCH + cc * 8;
#if ASYNC_LDS
        __builtin_amdgcn_global_load_async_to_lds_b128(
            (gp128)(gh + (size_t)chunk * 8), (lp128)(sBh + lofs), 0, 0);
        __builtin_amdgcn_global_load_async_to_lds_b128(
            (gp128)(gl + (size_t)chunk * 8), (lp128)(sBl + lofs), 0, 0);
#else
        *(v8h*)(sBh + lofs) = *(const v8h*)(gh + (size_t)chunk * 8);
        *(v8h*)(sBl + lofs) = *(const v8h*)(gl + (size_t)chunk * 8);
#endif
      }
    }
#if ASYNC_LDS
#if __has_builtin(__builtin_amdgcn_s_wait_asynccnt)
    __builtin_amdgcn_s_wait_asynccnt(0);
#else
    asm volatile("s_wait_asynccnt 0x0" ::: "memory");
#endif
#endif
    __syncthreads();

    v8f acc[4] = {};
#pragma unroll 2
    for (int kk = 0; kk < DDIM; kk += 32) {
      v16h Ah = join16(*(const v8h*)(xh_row + kk + aoff),
                       *(const v8h*)(xh_row + kk + aoff + 16));
      v16h Al = join16(*(const v8h*)(xl_row + kk + aoff),
                       *(const v8h*)(xl_row + kk + aoff + 16));
#pragma unroll
      for (int t = 0; t < 4; ++t) {
        const _Float16* bh = sBh + (t * 16 + nl) * LPITCH + bofs + kk;
        const _Float16* bl = sBl + (t * 16 + nl) * LPITCH + bofs + kk;
        v16h Bh = join16(*(const v8h*)bh, *(const v8h*)(bh + 16));
        v16h Bl = join16(*(const v8h*)bl, *(const v8h*)(bl + 16));
        acc[t] = wmma16(Ah, Bh, acc[t]);   // hi*hi
        acc[t] = wmma16(Ah, Bl, acc[t]);   // hi*lo
        acc[t] = wmma16(Al, Bh, acc[t]);   // lo*hi
      }
    }

    // Fused epilogue: phi-contribution = we[c] * exp2(2*sigma*log2e * dot).
    // C/D layout: VGPR j holds (M = j + 8*hiH, N = nl).
#pragma unroll
    for (int t = 0; t < 4; ++t) {
      float wec = we[cbase + t * 16 + nl];
#pragma unroll
      for (int j = 0; j < 8; ++j) {
        acc_out[j] = fmaf(wec, fast_exp2(s2 * acc[t][j]), acc_out[j]);
      }
    }
  }

  // Reduce the 16 lanes sharing each row; write per-K-slice partial.
#pragma unroll
  for (int j = 0; j < 8; ++j) {
    float v = acc_out[j];
    v += __shfl_xor(v, 1, 32);
    v += __shfl_xor(v, 2, 32);
    v += __shfl_xor(v, 4, 32);
    v += __shfl_xor(v, 8, 32);
    if (nl == 0)
      part[(size_t)blockIdx.y * NROWS + (size_t)(m0 + j + hiH * 8)] = v;
  }
}

extern "C" void kernel_launch(void* const* d_in, const int* in_sizes, int n_in,
                              void* d_out, int out_size, void* d_ws, size_t ws_size,
                              hipStream_t stream) {
  (void)in_sizes; (void)n_in; (void)out_size; (void)ws_size;
  const float* x   = (const float*)d_in[0];
  const float* cen = (const float*)d_in[1];
  const float* w   = (const float*)d_in[2];
  const float* sig = (const float*)d_in[3];
  float* out = (float*)d_out;

  // Workspace layout (~12.9 MB total)
  char* ws = (char*)d_ws;
  constexpr size_t SZ_XPLANE = (size_t)NROWS * DDIM * sizeof(_Float16); // 4 MiB
  constexpr size_t SZ_CPLANE = (size_t)KCENT * DDIM * sizeof(_Float16); // 2 MiB
  _Float16* xh  = (_Float16*)(ws);
  _Float16* xl  = (_Float16*)(ws + SZ_XPLANE);
  _Float16* chp = (_Float16*)(ws + 2 * SZ_XPLANE);
  _Float16* clp = (_Float16*)(ws + 2 * SZ_XPLANE + SZ_CPLANE);
  float* exp_ = (float*)(ws + 2 * SZ_XPLANE + 2 * SZ_CPLANE);
  float* wep  = (float*)((char*)exp_ + (size_t)NROWS * sizeof(float));
  float* partp = (float*)((char*)wep + (size_t)KCENT * sizeof(float));

  k_split<<<(NROWS * DDIM + 255) / 256, 256, 0, stream>>>(x, xh, xl, NROWS * DDIM);
  k_split<<<(KCENT * DDIM + 255) / 256, 256, 0, stream>>>(cen, chp, clp, KCENT * DDIM);
  k_prep_x<<<(NROWS + 127) / 128, 128, 0, stream>>>(x, exp_, sig, NROWS);
  k_prep_c<<<(KCENT + 127) / 128, 128, 0, stream>>>(cen, w, wep, sig, KCENT);

  dim3 grid(NROWS / 64, KCENT / 512);   // 128 x 8 = 1024 blocks
  k_rbf<<<grid, 128, 0, stream>>>(xh, xl, chp, clp, wep, sig, partp);

  k_reduce<<<(NROWS + 255) / 256, 256, 0, stream>>>(partp, exp_, out, NROWS,
                                                    KCENT / 512);
}
